// RetinaNet_12008728560059
// MI455X (gfx1250) — matrix operand order repaired and checked by reading, not compiled, not run on hardware
//
#include <hip/hip_runtime.h>
#include <hip/hip_bf16.h>
#include <math.h>

// ---------------------------------------------------------------------------
// CDNA5 / gfx1250 RetinaNet forward.
// Convs: double-buffered, async-LDS-staged implicit-GEMM on the WMMA pipe
// (v_wmma_f32_16x16x32_bf16, wave32, 8 waves/block, 64x128 tile, 2x2 register
// blocking per wave, global_load_async_to_lds staging overlapped with WMMA).
// Decode/NMS: VALU kernels.
// ---------------------------------------------------------------------------

typedef __bf16 bf16_t;
typedef bf16_t v16bf __attribute__((ext_vector_type(16)));
typedef float  v8f   __attribute__((ext_vector_type(8)));

#define NUM_CLASSES 91
#define NUM_ANCHORS 9
#define CAND_CAP    5000
#define OUT_KEEP    300
#define SCORE_TH    0.05f
#define IOU_TH      0.5f
#define BBOX_CLIP   4.135166556742356f   // log(1000/16)
#define LABEL_OFF   4096.0f

#define M_TILE 64
#define N_TILE 128

// ---- gfx1250 async global->LDS copies (ASYNCcnt) with sync fallback --------
#if defined(__HIP_DEVICE_COMPILE__) && \
    __has_builtin(__builtin_amdgcn_global_load_async_to_lds_b32) && \
    __has_builtin(__builtin_amdgcn_global_load_async_to_lds_b128)
#define HAVE_ASYNC_LDS 1
#endif

typedef int v4i_vs __attribute__((vector_size(16)));
typedef __attribute__((address_space(1))) int    as1_int;
typedef __attribute__((address_space(3))) int    as3_int;
typedef __attribute__((address_space(1))) v4i_vs as1_v4i;
typedef __attribute__((address_space(3))) v4i_vs as3_v4i;

__device__ __forceinline__ void copy_g2l_b32(const float* g, float* l) {
#ifdef HAVE_ASYNC_LDS
  __builtin_amdgcn_global_load_async_to_lds_b32(
      (as1_int*)(const void*)g, (as3_int*)(void*)l, 0, 0);
#else
  *l = *g;
#endif
}

__device__ __forceinline__ void copy_g2l_b128(const float* g, float* l) {
#ifdef HAVE_ASYNC_LDS
  __builtin_amdgcn_global_load_async_to_lds_b128(
      (as1_v4i*)(const void*)g, (as3_v4i*)(void*)l, 0, 0);
#else
  *(float4*)l = *(const float4*)g;
#endif
}

__device__ __forceinline__ void wait_async_lds() {
#ifdef HAVE_ASYNC_LDS
#if __has_builtin(__builtin_amdgcn_s_wait_asynccnt)
  __builtin_amdgcn_s_wait_asynccnt(0);
#else
  asm volatile("s_wait_asynccnt 0x0" ::: "memory");
#endif
#endif
}

// anchor (w,h) per level per anchor (BASES from the reference)
__constant__ float c_anchor_wh[5][NUM_ANCHORS][2] = {
  {{46,22},{56,28},{70,36},{32,32},{40,40},{50,50},{22,46},{28,56},{36,70}},
  {{90,46},{114,56},{142,72},{64,64},{80,80},{100,100},{46,90},{56,114},{72,142}},
  {{182,90},{228,114},{288,144},{128,128},{160,160},{204,204},{90,182},{114,228},{144,288}},
  {{362,182},{456,228},{574,288},{256,256},{322,322},{406,406},{182,362},{228,456},{288,574}},
  {{724,362},{912,456},{1148,574},{512,512},{644,644},{812,812},{362,724},{456,912},{574,1148}}
};

// pack two f32 -> two bf16 (round-to-nearest-even) in one dword
__device__ __forceinline__ unsigned pack2bf(float a, float b) {
  unsigned ua = __builtin_bit_cast(unsigned, a);
  unsigned ub = __builtin_bit_cast(unsigned, b);
  ua = ua + 0x7FFFu + ((ua >> 16) & 1u);
  ub = ub + 0x7FFFu + ((ub >> 16) & 1u);
  return (ua >> 16) | (ub & 0xFFFF0000u);
}

union FragU { uint4 q[2]; v16bf v; };

// build a 16-elem bf16 fragment from two runs of 8 consecutive f32 in LDS
__device__ __forceinline__ v16bf pack_frag(const float* r0, const float* r1) {
  float4 a0 = ((const float4*)r0)[0], a1 = ((const float4*)r0)[1];
  float4 b0 = ((const float4*)r1)[0], b1 = ((const float4*)r1)[1];
  FragU f;
  f.q[0] = make_uint4(pack2bf(a0.x, a0.y), pack2bf(a0.z, a0.w),
                      pack2bf(a1.x, a1.y), pack2bf(a1.z, a1.w));
  f.q[1] = make_uint4(pack2bf(b0.x, b0.y), pack2bf(b0.z, b0.w),
                      pack2bf(b1.x, b1.y), pack2bf(b1.z, b1.w));
  return f.v;
}

// ---------------------------------------------------------------------------
// Blocked implicit-GEMM conv (NCHW activations, OIHW weights).
//  - OIHW weights == row-major [Cout][Cin*KS*KS] (k = cin*KS*KS + ky*KS + kx)
//  - Double-buffered f32 LDS tiles: A 64x32 (async b128 rows), B 32x128
//    (per-lane async b32 im2col gather; padding lanes zero-fill via ds_store).
//  - 8 waves: wm=w>>2 (2x32 rows), wn=w&3 (4x32 cols); 2x2 accumulators per
//    wave -> 4 v_wmma_f32_16x16x32_bf16 per 32-deep K step, overlapped with
//    the async staging of the next K chunk.
// ---------------------------------------------------------------------------
template<int KS>
__global__ void __launch_bounds__(256)
conv_wmma_kernel(const float* __restrict__ X, const float* __restrict__ Wt,
                 const float* __restrict__ Bi, float* __restrict__ Y,
                 int Cin, int H, int W, int Cout, int Ho, int Wo,
                 int stride, int do_relu)
{
  __shared__ float As32[2][M_TILE * 32];   // 2 x 8 KB
  __shared__ float Bs32[2][N_TILE * 32];   // 2 x 16 KB

  const int t    = threadIdx.x;
  const int lane = t & 31;
  const int w    = t >> 5;
  const int hi   = lane >> 4;
  const int lr   = lane & 15;
  const int wm   = w >> 2;              // 0..1 -> 32-row sub-tile
  const int wn   = w & 3;               // 0..3 -> 32-col sub-tile

  const int b    = blockIdx.z;
  const int m0   = blockIdx.x * M_TILE;
  const int p0   = blockIdx.y * N_TILE;
  const int npix = Ho * Wo;
  const int pad  = KS >> 1;
  const int K    = Cin * KS * KS;       // always a multiple of 32 here

  // staging assignments
  const int arow = t & 63;              // A: row, 8 consecutive k per thread
  const int akb  = (t >> 6) * 8;
  const int bcol = t & 127;             // B: col (pixel), 16 consecutive k
  const int bkb  = (t >> 7) * 16;
  const int bp   = p0 + bcol;
  const int boy  = (bp < npix) ? (bp / Wo) : 0;
  const int box_ = (bp < npix) ? (bp % Wo) : 0;

  auto stage = [&](int kk, int buf) {
    // A (weights): contiguous f32 rows -> 2x async b128 per thread
    {
      const int gm = m0 + arow;
      float* ld = &As32[buf][arow * 32 + akb];
      if (gm < Cout) {
        const float* g = Wt + (size_t)gm * K + kk + akb;
        copy_g2l_b128(g, ld);
        copy_g2l_b128(g + 4, ld + 4);
      } else {
        ((float4*)ld)[0] = make_float4(0.f, 0.f, 0.f, 0.f);
        ((float4*)ld)[1] = make_float4(0.f, 0.f, 0.f, 0.f);
      }
    }
    // B (im2col gather): per-lane async b32; OOB -> zero-fill
    {
      float* ld = &Bs32[buf][bcol * 32 + bkb];
#pragma unroll
      for (int j = 0; j < 16; ++j) {
        const int k = kk + bkb + j;
        bool inb = false;
        const float* g = X;
        if (bp < npix && k < K) {
          const int cin = k / (KS * KS);
          const int r   = k - cin * (KS * KS);
          const int ky  = r / KS, kx = r - ky * KS;
          const int iy  = boy * stride + ky - pad;
          const int ix  = box_ * stride + kx - pad;
          if (iy >= 0 && iy < H && ix >= 0 && ix < W) {
            g = X + (((size_t)b * Cin + cin) * H + iy) * W + ix;
            inb = true;
          }
        }
        if (inb) copy_g2l_b32(g, &ld[j]);
        else     ld[j] = 0.f;
      }
    }
  };

  v8f acc[2][2];
#pragma unroll
  for (int i = 0; i < 2; ++i)
#pragma unroll
    for (int j = 0; j < 2; ++j)
      acc[i][j] = v8f{0.f,0.f,0.f,0.f,0.f,0.f,0.f,0.f};

  // software pipeline: stage chunk 0, then overlap stage(k+1) with compute(k)
  stage(0, 0);
  int buf = 0;
  for (int kk = 0; kk < K; kk += 32) {
    wait_async_lds();      // this wave's async copies into `buf` complete
    __syncthreads();       // all waves' staging of `buf` visible; prev compute done
    if (kk + 32 < K) stage(kk + 32, buf ^ 1);   // async, overlaps WMMA below

    // fragments (runs are contiguous f32 in LDS; 4x ds_load_b128 + pack each)
    v16bf fa[2], fb[2];
#pragma unroll
    for (int im = 0; im < 2; ++im) {
      const float* rowp = &As32[buf][(wm * 32 + im * 16 + lr) * 32];
      fa[im] = pack_frag(rowp + hi * 8, rowp + 16 + hi * 8);
    }
#pragma unroll
    for (int in = 0; in < 2; ++in) {
      const float* colp = &Bs32[buf][(wn * 32 + in * 16 + lr) * 32];
      fb[in] = pack_frag(colp + hi * 8, colp + 16 + hi * 8);
    }

#pragma unroll
    for (int im = 0; im < 2; ++im)
#pragma unroll
      for (int in = 0; in < 2; ++in)
        acc[im][in] = __builtin_amdgcn_wmma_f32_16x16x32_bf16(
            false, fa[im], false, fb[in], (short)0, acc[im][in], false, false);

    buf ^= 1;
  }

  // ---- epilogue: D VGPR r -> row M = r + 8*hi, col N = lane&15 ----
#pragma unroll
  for (int im = 0; im < 2; ++im) {
    const int mrow = m0 + wm * 32 + im * 16;
#pragma unroll
    for (int in = 0; in < 2; ++in) {
      const int pp = p0 + wn * 32 + in * 16 + lr;
#pragma unroll
      for (int r = 0; r < 8; ++r) {
        const int co = mrow + r + 8 * hi;
        if (co < Cout && pp < npix) {
          float val = acc[im][in][r] + Bi[co];
          if (do_relu) val = fmaxf(val, 0.f);
          Y[((size_t)b * Cout + co) * npix + pp] = val;
        }
      }
    }
  }
}

// ---------------------------------------------------------------------------
// FPN nearest-upsample + add:  dst[b,c,y,x] += src[b,c, y*Hs/Ht, x*Ws/Wt]
// ---------------------------------------------------------------------------
__global__ void upsample_add_kernel(float* __restrict__ dst,
                                    const float* __restrict__ src,
                                    int C, int Ht, int Wt, int Hs, int Ws, int B)
{
  size_t total = (size_t)B * C * Ht * Wt;
  for (size_t idx = blockIdx.x * (size_t)blockDim.x + threadIdx.x; idx < total;
       idx += (size_t)gridDim.x * blockDim.x) {
    int x = (int)(idx % Wt); size_t tt = idx / Wt;
    int y = (int)(tt % Ht);  tt /= Ht;
    int c = (int)(tt % C);   int b = (int)(tt / C);
    int sy = (int)(((long long)y * Hs) / Ht);
    int sx = (int)(((long long)x * Ws) / Wt);
    dst[idx] += src[(((size_t)b * C + c) * Hs + sy) * Ws + sx];
  }
}

__global__ void relu_copy_kernel(float* __restrict__ dst,
                                 const float* __restrict__ src, size_t n)
{
  for (size_t i = blockIdx.x * (size_t)blockDim.x + threadIdx.x; i < n;
       i += (size_t)gridDim.x * blockDim.x)
    dst[i] = fmaxf(src[i], 0.f);
}

__global__ void zero_int_kernel(int* p, int n)
{
  int i = blockIdx.x * blockDim.x + threadIdx.x;
  if (i < n) p[i] = 0;
}

// ---------------------------------------------------------------------------
// Decode: sigmoid threshold + box decode; append to per-image candidate list.
// Heads stay NCHW; channel c = a*task + t matches the reference's
// transpose(0,2,3,1)+reshape indexing.
// ---------------------------------------------------------------------------
__global__ void decode_kernel(const float* __restrict__ cls,
                              const float* __restrict__ reg,
                              int level, int H, int W, int B,
                              float* __restrict__ cboxes,
                              float* __restrict__ cscores,
                              int*   __restrict__ clabels,
                              int*   __restrict__ ccount,
                              float imgW, float imgH, float stride)
{
  const int npix = H * W;
  const long long total = (long long)B * npix * NUM_ANCHORS * NUM_CLASSES;
  for (long long idx = blockIdx.x * (long long)blockDim.x + threadIdx.x;
       idx < total; idx += (long long)gridDim.x * blockDim.x) {
    const int cl = (int)(idx % NUM_CLASSES);
    long long tt = idx / NUM_CLASSES;
    const int a = (int)(tt % NUM_ANCHORS); tt /= NUM_ANCHORS;
    const int p = (int)(tt % npix);
    const int b = (int)(tt / npix);

    const float logit =
        cls[((size_t)b * (NUM_ANCHORS * NUM_CLASSES) + a * NUM_CLASSES + cl)
            * npix + p];
    const float s = 1.f / (1.f + __expf(-logit));
    if (s <= SCORE_TH) continue;

    const int y = p / W, x = p % W;
    const float px = x * stride, py = y * stride;
    const float pw = c_anchor_wh[level][a][0];
    const float ph = c_anchor_wh[level][a][1];
    const float* rb = reg + ((size_t)b * (NUM_ANCHORS * 4) + a * 4) * npix + p;
    const float d0 = rb[0];
    const float d1 = rb[(size_t)npix];
    const float d2 = fminf(rb[2 * (size_t)npix], BBOX_CLIP);
    const float d3 = fminf(rb[3 * (size_t)npix], BBOX_CLIP);
    const float cx = px + d0 * pw, cy = py + d1 * ph;
    const float ww = pw * __expf(d2), hh = ph * __expf(d3);
    const float x1 = fminf(fmaxf(cx - 0.5f * ww, 0.f), imgW);
    const float y1 = fminf(fmaxf(cy - 0.5f * hh, 0.f), imgH);
    const float x2 = fminf(fmaxf(cx + 0.5f * ww, 0.f), imgW);
    const float y2 = fminf(fmaxf(cy + 0.5f * hh, 0.f), imgH);

    const int pos = atomicAdd(&ccount[b], 1);
    if (pos < CAND_CAP) {
      const size_t o = (size_t)b * CAND_CAP + pos;
      cboxes[o * 4 + 0] = x1; cboxes[o * 4 + 1] = y1;
      cboxes[o * 4 + 2] = x2; cboxes[o * 4 + 3] = y2;
      cscores[o] = s; clabels[o] = cl;
    }
  }
}

// ---------------------------------------------------------------------------
// Greedy class-aware NMS. One block per image; writes concatenated
// (boxes[B,300,4], scores[B,300], labels[B,300]).
// ---------------------------------------------------------------------------
__global__ void __launch_bounds__(256)
nms_kernel(const float* __restrict__ cboxes, const float* __restrict__ cscores,
           const int* __restrict__ clabels, const int* __restrict__ ccount,
           unsigned char* __restrict__ alive, float* __restrict__ out, int B)
{
  const int b = blockIdx.x;
  const int tid = threadIdx.x;
  int n = ccount[b]; if (n > CAND_CAP) n = CAND_CAP;

  const float* bx = cboxes + (size_t)b * CAND_CAP * 4;
  const float* sc = cscores + (size_t)b * CAND_CAP;
  const int*   lb = clabels + (size_t)b * CAND_CAP;
  unsigned char* al = alive + (size_t)b * CAND_CAP;

  for (int i = tid; i < CAND_CAP; i += blockDim.x) al[i] = (i < n) ? 1 : 0;
  __syncthreads();

  __shared__ float sv[256];
  __shared__ int   si[256];

  float* obox = out;
  float* osc  = out + (size_t)B * OUT_KEEP * 4;
  float* olb  = osc + (size_t)B * OUT_KEEP;

  for (int it = 0; it < OUT_KEEP; ++it) {
    float best = -1.f; int bi = -1;
    for (int i = tid; i < n; i += blockDim.x)
      if (al[i] && sc[i] > best) { best = sc[i]; bi = i; }
    sv[tid] = best; si[tid] = bi;
    __syncthreads();
    for (int s = 128; s > 0; s >>= 1) {
      if (tid < s && sv[tid + s] > sv[tid]) { sv[tid] = sv[tid + s]; si[tid] = si[tid + s]; }
      __syncthreads();
    }
    const int sel = si[0];
    const float selScore = sv[0];

    if (tid == 0) {
      const size_t o = (size_t)b * OUT_KEEP + it;
      if (sel >= 0) {
        obox[o * 4 + 0] = bx[sel * 4 + 0];
        obox[o * 4 + 1] = bx[sel * 4 + 1];
        obox[o * 4 + 2] = bx[sel * 4 + 2];
        obox[o * 4 + 3] = bx[sel * 4 + 3];
        osc[o] = selScore;
        olb[o] = (float)lb[sel];
        al[sel] = 0;
      } else {
        obox[o * 4 + 0] = 0.f; obox[o * 4 + 1] = 0.f;
        obox[o * 4 + 2] = 0.f; obox[o * 4 + 3] = 0.f;
        osc[o] = -1.f; olb[o] = 0.f;
      }
    }
    __syncthreads();

    if (sel >= 0) {
      const float off = lb[sel] * LABEL_OFF;
      const float X1 = bx[sel * 4 + 0] + off, Y1 = bx[sel * 4 + 1] + off;
      const float X2 = bx[sel * 4 + 2] + off, Y2 = bx[sel * 4 + 3] + off;
      const float areaS = (X2 - X1) * (Y2 - Y1);
      for (int i = tid; i < n; i += blockDim.x) {
        if (!al[i]) continue;
        const float o2 = lb[i] * LABEL_OFF;
        const float ax1 = bx[i * 4 + 0] + o2, ay1 = bx[i * 4 + 1] + o2;
        const float ax2 = bx[i * 4 + 2] + o2, ay2 = bx[i * 4 + 3] + o2;
        const float iw = fmaxf(fminf(X2, ax2) - fmaxf(X1, ax1), 0.f);
        const float ih = fmaxf(fminf(Y2, ay2) - fmaxf(Y1, ay1), 0.f);
        const float inter = iw * ih;
        const float iou =
            inter / ((areaS + (ax2 - ax1) * (ay2 - ay1) - inter) + 1e-9f);
        if (iou > IOU_TH) al[i] = 0;
      }
    }
    __syncthreads();
  }
}

// ---------------------------------------------------------------------------
// Host orchestration
// ---------------------------------------------------------------------------
extern "C" void kernel_launch(void* const* d_in, const int* in_sizes, int n_in,
                              void* d_out, int out_size, void* d_ws, size_t ws_size,
                              hipStream_t stream)
{
  (void)in_sizes; (void)n_in; (void)out_size; (void)ws_size;
  const int B = 2;
  const int Hs[5] = {100, 50, 25, 13, 7};
  const int Ws[5] = {152, 76, 38, 19, 10};
  size_t np[5]; for (int l = 0; l < 5; ++l) np[l] = (size_t)Hs[l] * Ws[l];
  const float strides[5] = {8.f, 16.f, 32.f, 64.f, 128.f};
  const float imgH = 800.f, imgW = 1216.f;

  // ---- inputs (setup_inputs dict order, params leaves in insertion order) --
  const float* C[3] = {(const float*)d_in[0], (const float*)d_in[1], (const float*)d_in[2]};
  const int Cin_lat[3] = {512, 1024, 2048};
  const int CH[3] = {100, 50, 25}, CW[3] = {152, 76, 38};
  int q = 3;
  const float* lat_w[3];   for (int i = 0; i < 3; ++i) lat_w[i]  = (const float*)d_in[q++];
  const float* lat_b[3];   for (int i = 0; i < 3; ++i) lat_b[i]  = (const float*)d_in[q++];
  const float* sm_w[3];    for (int i = 0; i < 3; ++i) sm_w[i]   = (const float*)d_in[q++];
  const float* sm_b[3];    for (int i = 0; i < 3; ++i) sm_b[i]   = (const float*)d_in[q++];
  const float* p6_w = (const float*)d_in[q++];
  const float* p6_b = (const float*)d_in[q++];
  const float* p7_w = (const float*)d_in[q++];
  const float* p7_b = (const float*)d_in[q++];
  const float* reg_cw[4];  for (int i = 0; i < 4; ++i) reg_cw[i] = (const float*)d_in[q++];
  const float* reg_cb[4];  for (int i = 0; i < 4; ++i) reg_cb[i] = (const float*)d_in[q++];
  const float* reg_fw = (const float*)d_in[q++];
  const float* reg_fb = (const float*)d_in[q++];
  const float* cls_cw[4];  for (int i = 0; i < 4; ++i) cls_cw[i] = (const float*)d_in[q++];
  const float* cls_cb[4];  for (int i = 0; i < 4; ++i) cls_cb[i] = (const float*)d_in[q++];
  const float* cls_fw = (const float*)d_in[q++];
  const float* cls_fb = (const float*)d_in[q++];

  // ---- workspace layout (floats) ----
  float* ws = (float*)d_ws;
  size_t off = 0;
  float* lat[3];  for (int l = 0; l < 3; ++l) { lat[l]  = ws + off; off += (size_t)B * 256 * np[l]; }
  float* feat[5]; for (int l = 0; l < 5; ++l) { feat[l] = ws + off; off += (size_t)B * 256 * np[l]; }
  float* relu6 = ws + off; off += (size_t)B * 256 * np[3];
  float* tmpA  = ws + off; off += (size_t)B * 256 * np[0];
  float* tmpB  = ws + off; off += (size_t)B * 256 * np[0];
  float* clsb[5]; for (int l = 0; l < 5; ++l) { clsb[l] = ws + off; off += (size_t)B * NUM_ANCHORS * NUM_CLASSES * np[l]; }
  float* regb[5]; for (int l = 0; l < 5; ++l) { regb[l] = ws + off; off += (size_t)B * NUM_ANCHORS * 4 * np[l]; }
  float* cboxes  = ws + off; off += (size_t)B * CAND_CAP * 4;
  float* cscores = ws + off; off += (size_t)B * CAND_CAP;
  int*   clabels = (int*)(ws + off); off += (size_t)B * CAND_CAP;
  int*   ccount  = (int*)(ws + off); off += 8;
  unsigned char* alive = (unsigned char*)(ws + off);

  auto conv = [&](const float* X, const float* Wt, const float* Bi, float* Y,
                  int Cin, int H, int W, int Cout, int Ho, int Wo,
                  int ks, int stride, int relu) {
    dim3 grid((Cout + M_TILE - 1) / M_TILE,
              (Ho * Wo + N_TILE - 1) / N_TILE, B);
    if (ks == 1)
      conv_wmma_kernel<1><<<grid, 256, 0, stream>>>(X, Wt, Bi, Y, Cin, H, W,
                                                    Cout, Ho, Wo, stride, relu);
    else
      conv_wmma_kernel<3><<<grid, 256, 0, stream>>>(X, Wt, Bi, Y, Cin, H, W,
                                                    Cout, Ho, Wo, stride, relu);
  };

  // ---- FPN ----
  for (int l = 2; l >= 0; --l)
    conv(C[l], lat_w[l], lat_b[l], lat[l], Cin_lat[l], CH[l], CW[l],
         256, CH[l], CW[l], 1, 1, 0);
  upsample_add_kernel<<<1024, 256, 0, stream>>>(lat[1], lat[2], 256, Hs[1], Ws[1], Hs[2], Ws[2], B);
  upsample_add_kernel<<<2048, 256, 0, stream>>>(lat[0], lat[1], 256, Hs[0], Ws[0], Hs[1], Ws[1], B);
  for (int l = 0; l < 3; ++l)
    conv(lat[l], sm_w[l], sm_b[l], feat[l], 256, Hs[l], Ws[l], 256, Hs[l], Ws[l], 3, 1, 0);
  conv(feat[2], p6_w, p6_b, feat[3], 256, Hs[2], Ws[2], 256, Hs[3], Ws[3], 3, 2, 0);
  relu_copy_kernel<<<256, 256, 0, stream>>>(relu6, feat[3], (size_t)B * 256 * np[3]);
  conv(relu6, p7_w, p7_b, feat[4], 256, Hs[3], Ws[3], 256, Hs[4], Ws[4], 3, 2, 0);

  // ---- heads ----
  for (int l = 0; l < 5; ++l) {
    const int H = Hs[l], W = Ws[l];
    const float* x = feat[l]; float* pa = tmpA; float* pb = tmpB;
    for (int j = 0; j < 4; ++j) {
      conv(x, reg_cw[j], reg_cb[j], pa, 256, H, W, 256, H, W, 3, 1, 1);
      x = pa; float* tswap = pa; pa = pb; pb = tswap;
    }
    conv(x, reg_fw, reg_fb, regb[l], 256, H, W, NUM_ANCHORS * 4, H, W, 3, 1, 0);
    x = feat[l]; pa = tmpA; pb = tmpB;
    for (int j = 0; j < 4; ++j) {
      conv(x, cls_cw[j], cls_cb[j], pa, 256, H, W, 256, H, W, 3, 1, 1);
      x = pa; float* tswap = pa; pa = pb; pb = tswap;
    }
    conv(x, cls_fw, cls_fb, clsb[l], 256, H, W, NUM_ANCHORS * NUM_CLASSES, H, W, 3, 1, 0);
  }

  // ---- decode + NMS ----
  zero_int_kernel<<<1, 32, 0, stream>>>(ccount, B);
  for (int l = 0; l < 5; ++l) {
    long long total = (long long)B * np[l] * NUM_ANCHORS * NUM_CLASSES;
    int blocks = (int)((total + 255) / 256);
    if (blocks > 8192) blocks = 8192;
    decode_kernel<<<blocks, 256, 0, stream>>>(clsb[l], regb[l], l, Hs[l], Ws[l],
                                              B, cboxes, cscores, clabels,
                                              ccount, imgW, imgH, strides[l]);
  }
  nms_kernel<<<B, 256, 0, stream>>>(cboxes, cscores, clabels, ccount, alive,
                                    (float*)d_out, B);
}
